// GraphPropLayer_21440476742360
// MI455X (gfx1250) — matrix operand order repaired and checked by reading, not compiled, not run on hardware
//
#include <hip/hip_runtime.h>
#include <hip/hip_bf16.h>
#include <math.h>

// ---------------------------------------------------------------------------
// GraphPropLayer for MI455X (gfx1250): fused edge-MLP + scatter, 3 GRUs.
// All GEMMs via v_wmma_f32_16x16x32_bf16 (f32 accumulate).
// M=32 per wave (each B-fragment feeds two WMMAs -> halves L2 weight traffic),
// register-cached A fragments, and double-buffered B fragments so ~12 weight
// loads stay in flight behind every WMMA chain (hides L2 latency at the
// LDS-limited occupancy of ~2 waves/SIMD).
// ---------------------------------------------------------------------------

typedef __bf16 bf16_t;
typedef __attribute__((ext_vector_type(16))) __bf16 v16bf;
typedef __attribute__((ext_vector_type(8)))  __bf16 v8bf;
typedef __attribute__((ext_vector_type(4)))  __bf16 v4bf;
typedef __attribute__((ext_vector_type(8)))  float  v8f;
typedef __attribute__((ext_vector_type(4)))  float  v4f;

#define DD 128   // node/edge feature dim
#define HH 256   // MLP hidden dim

static __device__ __forceinline__ v8f wmma_bf16(v16bf a, v16bf b, v8f c) {
  return __builtin_amdgcn_wmma_f32_16x16x32_bf16(false, a, false, b, (short)0, c,
                                                 false, false);
}

// A-matrix fragment (16x32 bf16) from an LDS tile stored row-major [rows][ktot].
// ISA 16-bit A layout: lane m=L&15, half=L>>4; elements 0..7 -> K=kt*32+half*8+e,
// elements 8..15 -> K=kt*32+16+half*8+(e-8). Two contiguous 16B chunks.
static __device__ __forceinline__ v16bf load_a_frag(const bf16_t* base, int ktot,
                                                    int m, int half, int kt) {
  const bf16_t* p = base + m * ktot + kt * 32 + half * 8;
  v8bf lo = *(const v8bf*)p;
  v8bf hi = *(const v8bf*)(p + 16);
  return __builtin_shufflevector(lo, hi, 0,1,2,3,4,5,6,7,8,9,10,11,12,13,14,15);
}

// B-matrix fragment: pre-packed so each lane's 16 bf16 are contiguous (32 B).
static __device__ __forceinline__ v16bf load_b_frag(const bf16_t* wf, int kt,
                                                    int nt, int NT, int lane) {
  return *(const v16bf*)(wf + ((long)(kt * NT + nt) * 32 + lane) * 16);
}

static __device__ __forceinline__ float sigmoidf_(float x) {
  return 1.0f / (1.0f + __expf(-x));
}

// ---------------------------------------------------------------------------
// Weight repack: w is [O, I] row-major f32 (y = x @ w^T). Produce bf16 B-tile
// fragments: tile (kt, nt), lane L (half=L>>4, nl=L&15), element e:
//   K = kt*32 + half*8 + e (e<8), kt*32 + 16 + half*8 + (e-8) (e>=8),
//   N = nt*16 + nl, value = w[N][K].
// ---------------------------------------------------------------------------
__global__ void pack_b_kernel(const float* __restrict__ w, bf16_t* __restrict__ outp,
                              int O, int I) {
  long idx = (long)blockIdx.x * 256 + threadIdx.x;
  long total = (long)O * I;
  if (idx >= total) return;
  int e    = (int)(idx & 15);
  int lane = (int)((idx >> 4) & 31);
  long frag = idx >> 9;
  int NT = O >> 4;
  int nt = (int)(frag % NT);
  int kt = (int)(frag / NT);
  int half = lane >> 4, nl = lane & 15;
  int K = kt * 32 + (e < 8 ? half * 8 + e : 16 + half * 8 + (e - 8));
  int Nc = nt * 16 + nl;
  outp[idx] = (bf16_t)w[(long)Nc * I + K];
}

__global__ void zero_kernel(float* __restrict__ p, int n) {
  int i = blockIdx.x * 256 + threadIdx.x;
  if (i < n) p[i] = 0.0f;
}

// ---------------------------------------------------------------------------
// Fused edge pipeline, one wave per 32-edge tile:
//   gather -> bf16 LDS [32 x 384] -> L1 (ReLU) -> L2 (ReLU) -> L3 ->
//   f32 atomic scatter-add into agg[N,128].
// Instance i < E: concat(ns[f[i]], ns[t[i]], ef[i]) scattered to t[i];
// instance i >= E: concat(ns[t[j]], ns[f[j]], ef[j]) scattered to f[j].
// LDS: X tile (24 KB) is dead after layer-1 A fragments are cached, so H2
// aliases it (LDS ops are in-order within a wave).
// ---------------------------------------------------------------------------
__global__ void __launch_bounds__(32) edge_mlp_kernel(
    const float* __restrict__ ns, const float* __restrict__ ef,
    const int* __restrict__ fidx, const int* __restrict__ tidx,
    const bf16_t* __restrict__ w1f, const float* __restrict__ b1,
    const bf16_t* __restrict__ w2f, const float* __restrict__ b2,
    const bf16_t* __restrict__ w3f, const float* __restrict__ b3,
    float* __restrict__ agg, int E) {
  __shared__ __align__(16) bf16_t sX [32 * 384];  // 24 KB; reused as H2 later
  __shared__ __align__(16) bf16_t sH1[32 * 256];  // 16 KB
  __shared__ int sTgt[32];

  const int lane = threadIdx.x & 31;
  const long inst0 = (long)blockIdx.x * 32;
  const long total = 2L * (long)E;
  const bool valid = inst0 < total;

  bf16_t* X  = sX;
  bf16_t* H1 = sH1;
  bf16_t* H2 = sX;   // alias: X dead once layer-1 A fragments are in registers

  // ---- gather + f32->bf16 convert into LDS
  const int c = lane * 4;  // 4 contiguous cols per lane per 128-wide segment
  for (int m = 0; m < 32; ++m) {
    long inst = inst0 + m;
    if (inst >= total) inst = total - 1;
    int e, s0, s1i, tg;
    if (inst < (long)E) { e = (int)inst;        s0 = fidx[e]; s1i = tidx[e]; tg = tidx[e]; }
    else                { e = (int)(inst - E);  s0 = tidx[e]; s1i = fidx[e]; tg = fidx[e]; }
    if (lane == 0) sTgt[m] = tg;
    v4f a0 = *(const v4f*)(ns + (long)s0  * DD + c);
    v4f a1 = *(const v4f*)(ns + (long)s1i * DD + c);
    // edge features are streaming (2*328MB > L2): non-temporal
    v4f a2 = __builtin_nontemporal_load((const v4f*)(ef + (long)e * DD + c));
    v4bf p0, p1, p2;
#pragma unroll
    for (int i = 0; i < 4; ++i) {
      p0[i] = (bf16_t)a0[i]; p1[i] = (bf16_t)a1[i]; p2[i] = (bf16_t)a2[i];
    }
    bf16_t* xr = X + m * 384;
    *(v4bf*)(xr + c)       = p0;
    *(v4bf*)(xr + 128 + c) = p1;
    *(v4bf*)(xr + 256 + c) = p2;
  }
  __syncthreads();  // single-wave workgroup: lowers to S_NOP, kept for clarity

  const int half = lane >> 4, nl = lane & 15, mA = lane & 15;
  const int mrow = half * 8;  // C/D layout: row = half*8 + r, col = nl

  // ---- layer 1: [32,384] @ [384,256]^T, ReLU -> H1 (B double-buffered)
  {
    v16bf a0[12], a1[12];
#pragma unroll
    for (int kt = 0; kt < 12; ++kt) {
      a0[kt] = load_a_frag(X, 384, mA,      half, kt);
      a1[kt] = load_a_frag(X, 384, mA + 16, half, kt);
    }
    v16bf bA[12], bB[12];
#pragma unroll
    for (int kt = 0; kt < 12; ++kt) bA[kt] = load_b_frag(w1f, kt, 0, 16, lane);
#pragma unroll
    for (int nt = 0; nt < 16; ++nt) {
      v16bf* bc = (nt & 1) ? bB : bA;
      v16bf* bn = (nt & 1) ? bA : bB;
      if (nt < 15) {
#pragma unroll
        for (int kt = 0; kt < 12; ++kt) bn[kt] = load_b_frag(w1f, kt, nt + 1, 16, lane);
      }
      float bias = b1[nt * 16 + nl];
      v8f acc0, acc1;
#pragma unroll
      for (int r = 0; r < 8; ++r) { acc0[r] = bias; acc1[r] = bias; }
#pragma unroll
      for (int kt = 0; kt < 12; ++kt) {
        acc0 = wmma_bf16(a0[kt], bc[kt], acc0);
        acc1 = wmma_bf16(a1[kt], bc[kt], acc1);
      }
#pragma unroll
      for (int r = 0; r < 8; ++r) {
        H1[(mrow + r)      * 256 + nt * 16 + nl] = (bf16_t)fmaxf(acc0[r], 0.0f);
        H1[(mrow + r + 16) * 256 + nt * 16 + nl] = (bf16_t)fmaxf(acc1[r], 0.0f);
      }
    }
  }

  // ---- layer 2: [32,256] @ [256,256]^T, ReLU -> H2 (aliases X)
  {
    v16bf a0[8], a1[8];
#pragma unroll
    for (int kt = 0; kt < 8; ++kt) {
      a0[kt] = load_a_frag(H1, 256, mA,      half, kt);
      a1[kt] = load_a_frag(H1, 256, mA + 16, half, kt);
    }
    v16bf bA[8], bB[8];
#pragma unroll
    for (int kt = 0; kt < 8; ++kt) bA[kt] = load_b_frag(w2f, kt, 0, 16, lane);
#pragma unroll
    for (int nt = 0; nt < 16; ++nt) {
      v16bf* bc = (nt & 1) ? bB : bA;
      v16bf* bn = (nt & 1) ? bA : bB;
      if (nt < 15) {
#pragma unroll
        for (int kt = 0; kt < 8; ++kt) bn[kt] = load_b_frag(w2f, kt, nt + 1, 16, lane);
      }
      float bias = b2[nt * 16 + nl];
      v8f acc0, acc1;
#pragma unroll
      for (int r = 0; r < 8; ++r) { acc0[r] = bias; acc1[r] = bias; }
#pragma unroll
      for (int kt = 0; kt < 8; ++kt) {
        acc0 = wmma_bf16(a0[kt], bc[kt], acc0);
        acc1 = wmma_bf16(a1[kt], bc[kt], acc1);
      }
#pragma unroll
      for (int r = 0; r < 8; ++r) {
        H2[(mrow + r)      * 256 + nt * 16 + nl] = (bf16_t)fmaxf(acc0[r], 0.0f);
        H2[(mrow + r + 16) * 256 + nt * 16 + nl] = (bf16_t)fmaxf(acc1[r], 0.0f);
      }
    }
  }

  // ---- layer 3: [32,256] @ [256,128]^T -> scatter-add (segment sum)
  {
    v16bf a0[8], a1[8];
#pragma unroll
    for (int kt = 0; kt < 8; ++kt) {
      a0[kt] = load_a_frag(H2, 256, mA,      half, kt);
      a1[kt] = load_a_frag(H2, 256, mA + 16, half, kt);
    }
    v16bf bA[8], bB[8];
#pragma unroll
    for (int kt = 0; kt < 8; ++kt) bA[kt] = load_b_frag(w3f, kt, 0, 8, lane);
#pragma unroll
    for (int nt = 0; nt < 8; ++nt) {
      v16bf* bc = (nt & 1) ? bB : bA;
      v16bf* bn = (nt & 1) ? bA : bB;
      if (nt < 7) {
#pragma unroll
        for (int kt = 0; kt < 8; ++kt) bn[kt] = load_b_frag(w3f, kt, nt + 1, 8, lane);
      }
      float bias = b3[nt * 16 + nl];
      v8f acc0, acc1;
#pragma unroll
      for (int r = 0; r < 8; ++r) { acc0[r] = bias; acc1[r] = bias; }
#pragma unroll
      for (int kt = 0; kt < 8; ++kt) {
        acc0 = wmma_bf16(a0[kt], bc[kt], acc0);
        acc1 = wmma_bf16(a1[kt], bc[kt], acc1);
      }
      if (valid) {
#pragma unroll
        for (int r = 0; r < 8; ++r) {
          int n0 = sTgt[mrow + r];
          int n1 = sTgt[mrow + r + 16];
          __hip_atomic_fetch_add(agg + (long)n0 * DD + nt * 16 + nl, acc0[r],
                                 __ATOMIC_RELAXED, __HIP_MEMORY_SCOPE_AGENT);
          __hip_atomic_fetch_add(agg + (long)n1 * DD + nt * 16 + nl, acc1[r],
                                 __ATOMIC_RELAXED, __HIP_MEMORY_SCOPE_AGENT);
        }
      }
    }
  }
}

// ---------------------------------------------------------------------------
// GRU cell, one wave per 32-node tile: out = (1-z)*tanh(in + r*hn) + z*h.
// wih/whh are [384,128]; K=128 (4 k-tiles), N=384 (24 n-tiles). For output
// column block j (0..7), gates need n-tiles j (r), j+8 (z), j+16 (n) of both
// gi and gh. All 24 B fragments of a j-iteration are loaded in one phase so
// up to 24 loads are in flight behind the 48-WMMA chain.
// ---------------------------------------------------------------------------
__global__ void __launch_bounds__(32) gru_kernel(
    const float* __restrict__ x, const float* h,
    const bf16_t* __restrict__ wihf, const bf16_t* __restrict__ whhf,
    const float* __restrict__ bih, const float* __restrict__ bhh,
    float* out, int Nn) {
  __shared__ __align__(16) bf16_t sXb[32 * 128];  // 8 KB
  __shared__ __align__(16) bf16_t sHb[32 * 128];  // 8 KB
  __shared__ __align__(16) float  sHf[32 * 128];  // 16 KB

  const int lane = threadIdx.x & 31;
  const int row0 = blockIdx.x * 32;
  const bool valid = row0 < Nn;

  const int c = lane * 4;
  for (int m = 0; m < 32; ++m) {
    int rr = row0 + m; if (rr >= Nn) rr = Nn - 1;
    v4f xv = *(const v4f*)(x + (long)rr * DD + c);
    v4f hv = *(const v4f*)(h + (long)rr * DD + c);
    v4bf xb, hb;
#pragma unroll
    for (int i = 0; i < 4; ++i) { xb[i] = (bf16_t)xv[i]; hb[i] = (bf16_t)hv[i]; }
    *(v4bf*)(sXb + m * 128 + c) = xb;
    *(v4bf*)(sHb + m * 128 + c) = hb;
    *(v4f*)(sHf + m * 128 + c)  = hv;
  }
  __syncthreads();

  const int half = lane >> 4, nl = lane & 15, mA = lane & 15;
  const int mrow = half * 8;

  // Cache A fragments for both 16-row tiles of x and h (K=128 -> 4 k-tiles).
  v16bf ax0[4], ax1[4], ah0[4], ah1[4];
#pragma unroll
  for (int kt = 0; kt < 4; ++kt) {
    ax0[kt] = load_a_frag(sXb, 128, mA,      half, kt);
    ax1[kt] = load_a_frag(sXb, 128, mA + 16, half, kt);
    ah0[kt] = load_a_frag(sHb, 128, mA,      half, kt);
    ah1[kt] = load_a_frag(sHb, 128, mA + 16, half, kt);
  }

#pragma unroll
  for (int j = 0; j < 8; ++j) {
    const int col = j * 16 + nl;
    // load phase: all 24 B fragments for this j
    v16bf bb[24];
#pragma unroll
    for (int kt = 0; kt < 4; ++kt) {
      bb[kt * 6 + 0] = load_b_frag(wihf, kt, j,      24, lane);
      bb[kt * 6 + 1] = load_b_frag(wihf, kt, j + 8,  24, lane);
      bb[kt * 6 + 2] = load_b_frag(wihf, kt, j + 16, 24, lane);
      bb[kt * 6 + 3] = load_b_frag(whhf, kt, j,      24, lane);
      bb[kt * 6 + 4] = load_b_frag(whhf, kt, j + 8,  24, lane);
      bb[kt * 6 + 5] = load_b_frag(whhf, kt, j + 16, 24, lane);
    }
    float br = bih[col], bz = bih[128 + col], bn = bih[256 + col];
    float cr = bhh[col], cz = bhh[128 + col], cn = bhh[256 + col];
    v8f gir0, giz0, gin0, ghr0, ghz0, ghn0;
    v8f gir1, giz1, gin1, ghr1, ghz1, ghn1;
#pragma unroll
    for (int r = 0; r < 8; ++r) {
      gir0[r] = br; giz0[r] = bz; gin0[r] = bn;
      ghr0[r] = cr; ghz0[r] = cz; ghn0[r] = cn;
      gir1[r] = br; giz1[r] = bz; gin1[r] = bn;
      ghr1[r] = cr; ghz1[r] = cz; ghn1[r] = cn;
    }
#pragma unroll
    for (int kt = 0; kt < 4; ++kt) {
      gir0 = wmma_bf16(ax0[kt], bb[kt*6+0], gir0);  gir1 = wmma_bf16(ax1[kt], bb[kt*6+0], gir1);
      giz0 = wmma_bf16(ax0[kt], bb[kt*6+1], giz0);  giz1 = wmma_bf16(ax1[kt], bb[kt*6+1], giz1);
      gin0 = wmma_bf16(ax0[kt], bb[kt*6+2], gin0);  gin1 = wmma_bf16(ax1[kt], bb[kt*6+2], gin1);
      ghr0 = wmma_bf16(ah0[kt], bb[kt*6+3], ghr0);  ghr1 = wmma_bf16(ah1[kt], bb[kt*6+3], ghr1);
      ghz0 = wmma_bf16(ah0[kt], bb[kt*6+4], ghz0);  ghz1 = wmma_bf16(ah1[kt], bb[kt*6+4], ghz1);
      ghn0 = wmma_bf16(ah0[kt], bb[kt*6+5], ghn0);  ghn1 = wmma_bf16(ah1[kt], bb[kt*6+5], ghn1);
    }
    if (valid) {
#pragma unroll
      for (int r = 0; r < 8; ++r) {
        int m0 = mrow + r;
        int m1 = mrow + r + 16;
        if (row0 + m0 < Nn) {
          float rg = sigmoidf_(gir0[r] + ghr0[r]);
          float zg = sigmoidf_(giz0[r] + ghz0[r]);
          float nn = tanhf(gin0[r] + rg * ghn0[r]);
          float hv = sHf[m0 * 128 + col];
          out[(long)(row0 + m0) * DD + col] = (1.0f - zg) * nn + zg * hv;
        }
        if (row0 + m1 < Nn) {
          float rg = sigmoidf_(gir1[r] + ghr1[r]);
          float zg = sigmoidf_(giz1[r] + ghz1[r]);
          float nn = tanhf(gin1[r] + rg * ghn1[r]);
          float hv = sHf[m1 * 128 + col];
          out[(long)(row0 + m1) * DD + col] = (1.0f - zg) * nn + zg * hv;
        }
      }
    }
  }
}

// ---------------------------------------------------------------------------
extern "C" void kernel_launch(void* const* d_in, const int* in_sizes, int n_in,
                              void* d_out, int out_size, void* d_ws, size_t ws_size,
                              hipStream_t stream) {
  (void)n_in; (void)out_size; (void)ws_size;
  const float* ns   = (const float*)d_in[0];
  const float* ef   = (const float*)d_in[1];
  const int*   fidx = (const int*)d_in[2];
  const int*   tidx = (const int*)d_in[3];
  const float* w1 = (const float*)d_in[5];
  const float* b1 = (const float*)d_in[6];
  const float* w2 = (const float*)d_in[7];
  const float* b2 = (const float*)d_in[8];
  const float* w3 = (const float*)d_in[9];
  const float* b3 = (const float*)d_in[10];
  const float *gwih[3], *gwhh[3], *gbih[3], *gbhh[3];
  for (int i = 0; i < 3; ++i) {
    gwih[i] = (const float*)d_in[11 + 4 * i];
    gwhh[i] = (const float*)d_in[12 + 4 * i];
    gbih[i] = (const float*)d_in[13 + 4 * i];
    gbhh[i] = (const float*)d_in[14 + 4 * i];
  }
  const int Nn = in_sizes[0] / DD;   // 100000
  const int Ee = in_sizes[2];        // 640000

  char* ws = (char*)d_ws;
  size_t off = 0;
  float* agg = (float*)(ws + off); off += (size_t)Nn * DD * 4;
  float* s1  = (float*)(ws + off); off += (size_t)Nn * DD * 4;
  bf16_t* w1f = (bf16_t*)(ws + off); off += (size_t)384 * 256 * 2;
  bf16_t* w2f = (bf16_t*)(ws + off); off += (size_t)256 * 256 * 2;
  bf16_t* w3f = (bf16_t*)(ws + off); off += (size_t)256 * 128 * 2;
  bf16_t *gwihf[3], *gwhhf[3];
  for (int i = 0; i < 3; ++i) {
    gwihf[i] = (bf16_t*)(ws + off); off += (size_t)128 * 384 * 2;
    gwhhf[i] = (bf16_t*)(ws + off); off += (size_t)128 * 384 * 2;
  }

  // 1) zero the aggregation buffer
  {
    int tot = Nn * DD;
    zero_kernel<<<(tot + 255) / 256, 256, 0, stream>>>(agg, tot);
  }
  // 2) repack weights to fragment-ready bf16
  auto pack = [&](const float* w, bf16_t* o, int O, int I) {
    int tot = O * I;
    pack_b_kernel<<<(tot + 255) / 256, 256, 0, stream>>>(w, o, O, I);
  };
  pack(w1, w1f, 256, 384);
  pack(w2, w2f, 256, 256);
  pack(w3, w3f, 128, 256);
  for (int i = 0; i < 3; ++i) {
    pack(gwih[i], gwihf[i], 384, 128);
    pack(gwhh[i], gwhhf[i], 384, 128);
  }
  // 3) fused edge MLP + scatter (both directions): 2E instances, 32 per wave
  {
    long tiles = (2L * Ee + 31) / 32;
    edge_mlp_kernel<<<(int)tiles, 32, 0, stream>>>(ns, ef, fidx, tidx,
                                                   w1f, b1, w2f, b2, w3f, b3,
                                                   agg, Ee);
  }
  // 4) GRU chain: s1 = GRU0(x=ns, h=agg); s2 = GRU1(x=agg, h=s1) -> d_out;
  //    s3 = GRU2(x=s1, h=s2) -> d_out (in place; each block reads its own
  //    rows into LDS before writing them).
  {
    int tiles = (Nn + 31) / 32;
    float* outp = (float*)d_out;
    gru_kernel<<<tiles, 32, 0, stream>>>(ns,  agg, gwihf[0], gwhhf[0],
                                         gbih[0], gbhh[0], s1, Nn);
    gru_kernel<<<tiles, 32, 0, stream>>>(agg, s1,  gwihf[1], gwhhf[1],
                                         gbih[1], gbhh[1], outp, Nn);
    gru_kernel<<<tiles, 32, 0, stream>>>(s1,  outp, gwihf[2], gwhhf[2],
                                         gbih[2], gbhh[2], outp, Nn);
  }
}
